// Model_20547123544441
// MI455X (gfx1250) — matrix operand order repaired and checked by reading, not compiled, not run on hardware
//
#include <hip/hip_runtime.h>

// ---------------------------------------------------------------------------
// Fire module (SqueezeNet) for MI455X / gfx1250, wave32 + WMMA bf16.
//   squeeze 1x1 (512->64) -> s (bf16, workspace)
//   expand 1x1 (64->256)  -> out[:,0:256]
//   expand 3x3 (64->256)  -> out[:,256:512]   (halo-in-LDS, fill once/block)
// All matrix math via v_wmma_f32_16x16x32_bf16, fp32 accumulate.
// B-fragment loads batched ahead of WMMA groups so LDS reads clause together.
// ---------------------------------------------------------------------------

typedef __bf16 bf16_t;
typedef __attribute__((ext_vector_type(16))) __bf16 v16bf;
typedef __attribute__((ext_vector_type(8)))  __bf16 v8bf;
typedef __attribute__((ext_vector_type(8)))  float  v8f;

union FragU { v16bf v; v8bf h[2]; };

#define C_IN   512
#define C_SQ   64
#define C_EX   256
#define HW     3136
#define IMG_W  56
#define NIMG   32
#define NT     128   // pixel-tile (GEMM N) for squeeze / expand1x1
#define BSTR   40    // LDS B row stride, bf16 elems (80B: 16B-aligned)

// expand3x3 halo tile: 4 output rows -> 6 input rows x 58 cols x 64 ch
#define E3_ROWS 4
#define HROWS   (E3_ROWS + 2)        // 6
#define HCOLS   (IMG_W + 2)          // 58
#define HSTR    72                   // LDS halo row stride (144B, 16B-aligned)

__device__ __forceinline__ unsigned bfbits(float f) {
  unsigned u = __float_as_uint(f);
  u += 0x7FFFu + ((u >> 16) & 1u);           // round-to-nearest-even
  return u >> 16;
}

__device__ __forceinline__ bf16_t f2bf(float f) {
  unsigned short h = (unsigned short)bfbits(f);
  return __builtin_bit_cast(bf16_t, h);
}

__device__ __forceinline__ v8f wmma_bf16(v16bf a, v16bf b, v8f c) {
  return __builtin_amdgcn_wmma_f32_16x16x32_bf16(
      false, a, false, b, (short)0, c, false, false);
}

// A-matrix (16x32, 16-bit): lane half selects K-interleave of 8 (ISA 7.12.2)
__device__ __forceinline__ v16bf load_fragA(const bf16_t* row, int lane) {
  const int ko = (lane & 16) ? 8 : 0;
  FragU f;
  f.h[0] = *(const v8bf*)(row + ko);
  f.h[1] = *(const v8bf*)(row + ko + 16);
  return f.v;
}

// B-matrix (32x16, 16-bit): lane half selects K-interleave of 8 (matches A)
__device__ __forceinline__ v16bf load_fragB(const bf16_t* row, int lane) {
  const int ko = (lane & 16) ? 8 : 0;
  FragU f;
  f.h[0] = *(const v8bf*)(row + ko);
  f.h[1] = *(const v8bf*)(row + ko + 16);
  return f.v;
}

// ---------------------------------------------------------------------------
// Weight prep: fp32 -> bf16; w3x3 reordered to [tap(9)][o(256)][i(64)]
// ---------------------------------------------------------------------------
__global__ __launch_bounds__(256) void k_prep(
    const float* __restrict__ wsq, const float* __restrict__ w1,
    const float* __restrict__ w3,
    bf16_t* __restrict__ o_wsq, bf16_t* __restrict__ o_w1,
    bf16_t* __restrict__ o_w3)
{
  int i = blockIdx.x * 256 + threadIdx.x;
  if (i < C_SQ * C_IN) o_wsq[i] = f2bf(wsq[i]);
  if (i < C_EX * C_SQ) o_w1[i] = f2bf(w1[i]);
  if (i < C_EX * C_SQ * 9) {
    int t = i % 9;              // ky*3+kx
    int rest = i / 9;
    int c = rest % C_SQ;
    int o = rest / C_SQ;
    o_w3[((size_t)t * C_EX + o) * C_SQ + c] = f2bf(w3[i]);
  }
}

// ---------------------------------------------------------------------------
// Squeeze: s[n][c][p] = relu(Wsq(64x512) @ x[n](512x3136) + b)
// Block: 64(M) x 128(N); wave -> 1 M-tile x 4 N-tiles.
// ---------------------------------------------------------------------------
__global__ __launch_bounds__(256) void k_squeeze(
    const float* __restrict__ x, const bf16_t* __restrict__ wsq,
    const float* __restrict__ bsq, bf16_t* __restrict__ s)
{
  __shared__ __align__(16) bf16_t ldsB[NT * BSTR];   // [pixel][k]
  const int tid = threadIdx.x;
  const int lane = tid & 31, wave = tid >> 5;
  const int lrow = lane & 15;
  const int hi8 = (lane & 16) ? 8 : 0;
  const int mtile = wave & 3;
  const int ngrp = wave >> 2;
  const int qb = blockIdx.x * NT;
  const int n0 = qb / HW;
  const int p0 = qb - n0 * HW;

  v8f acc[4];
  #pragma unroll
  for (int j = 0; j < 4; ++j)
    #pragma unroll
    for (int e = 0; e < 8; ++e) acc[j][e] = 0.0f;

  for (int kb = 0; kb < C_IN; kb += 32) {
    __syncthreads();
    #pragma unroll
    for (int i = 0; i < 8; ++i) {          // 2048 channel-pairs / 256 thr
      int idx = tid + i * 256;
      int c = (idx >> 7) * 2;              // even channel 0..30
      int p = idx & 127;
      int pp = p0 + p, n = n0;
      if (pp >= HW) { ++n; pp -= HW; }
      const float* px = x + ((size_t)n * C_IN + kb + c) * HW + pp;
      unsigned pk = bfbits(px[0]) | (bfbits(px[HW]) << 16);
      *(unsigned*)&ldsB[p * BSTR + c] = pk;
    }
    __syncthreads();

    v16bf afrag = load_fragA(wsq + (size_t)(mtile * 16 + lrow) * C_IN + kb, lane);
    v16bf bfr[4];
    #pragma unroll
    for (int j = 0; j < 4; ++j)
      bfr[j] = load_fragB(&ldsB[((ngrp * 4 + j) * 16 + lrow) * BSTR], lane);
    #pragma unroll
    for (int j = 0; j < 4; ++j)
      acc[j] = wmma_bf16(afrag, bfr[j], acc[j]);
  }

  #pragma unroll
  for (int j = 0; j < 4; ++j) {
    int pl = (ngrp * 4 + j) * 16 + lrow;
    int pp = p0 + pl, n = n0;
    if (pp >= HW) { ++n; pp -= HW; }
    #pragma unroll
    for (int r = 0; r < 8; ++r) {
      int c = mtile * 16 + r + hi8;
      float v = acc[j][r] + bsq[c];
      v = v > 0.0f ? v : 0.0f;
      s[((size_t)n * C_SQ + c) * HW + pp] = f2bf(v);
    }
  }
}

// ---------------------------------------------------------------------------
// Expand 1x1: out[:,0:256] = relu(W1(256x64) @ s + b1). Grid.y = M-block/64.
// ---------------------------------------------------------------------------
__global__ __launch_bounds__(256) void k_expand1(
    const bf16_t* __restrict__ s, const bf16_t* __restrict__ w1,
    const float* __restrict__ b1, float* __restrict__ out)
{
  __shared__ __align__(16) bf16_t ldsB[NT * BSTR];
  const int tid = threadIdx.x;
  const int lane = tid & 31, wave = tid >> 5;
  const int lrow = lane & 15;
  const int hi8 = (lane & 16) ? 8 : 0;
  const int mtile = wave & 3;
  const int ngrp = wave >> 2;
  const int qb = blockIdx.x * NT;
  const int cb = blockIdx.y * 64;
  const int n0 = qb / HW;
  const int p0 = qb - n0 * HW;

  v8f acc[4];
  #pragma unroll
  for (int j = 0; j < 4; ++j)
    #pragma unroll
    for (int e = 0; e < 8; ++e) acc[j][e] = 0.0f;

  for (int kb = 0; kb < C_SQ; kb += 32) {
    __syncthreads();
    #pragma unroll
    for (int i = 0; i < 8; ++i) {          // packed channel-pairs
      int idx = tid + i * 256;
      int c = (idx >> 7) * 2;
      int p = idx & 127;
      int pp = p0 + p, n = n0;
      if (pp >= HW) { ++n; pp -= HW; }
      const unsigned short* sp =
          (const unsigned short*)s + ((size_t)n * C_SQ + kb + c) * HW + pp;
      unsigned pk = (unsigned)sp[0] | ((unsigned)sp[HW] << 16);
      *(unsigned*)&ldsB[p * BSTR + c] = pk;
    }
    __syncthreads();

    v16bf afrag = load_fragA(w1 + (size_t)(cb + mtile * 16 + lrow) * C_SQ + kb, lane);
    v16bf bfr[4];
    #pragma unroll
    for (int j = 0; j < 4; ++j)
      bfr[j] = load_fragB(&ldsB[((ngrp * 4 + j) * 16 + lrow) * BSTR], lane);
    #pragma unroll
    for (int j = 0; j < 4; ++j)
      acc[j] = wmma_bf16(afrag, bfr[j], acc[j]);
  }

  #pragma unroll
  for (int j = 0; j < 4; ++j) {
    int pl = (ngrp * 4 + j) * 16 + lrow;
    int pp = p0 + pl, n = n0;
    if (pp >= HW) { ++n; pp -= HW; }
    #pragma unroll
    for (int r = 0; r < 8; ++r) {
      int c = cb + mtile * 16 + r + hi8;                // 0..255
      float v = acc[j][r] + b1[c];
      v = v > 0.0f ? v : 0.0f;
      out[((size_t)n * 512 + c) * HW + pp] = v;
    }
  }
}

// ---------------------------------------------------------------------------
// Expand 3x3 (pad 1): out[:,256:512].
// Block: n, 4 output rows (N=224 = 14 N-tiles), 64 output channels.
// Halo of s (6x58 pixels x 64 ch, zero-padded) loaded to LDS ONCE, then
// 9 taps x 2 K-chunks of WMMA with no further syncs/fills.
// ---------------------------------------------------------------------------
__global__ __launch_bounds__(256) void k_expand3(
    const bf16_t* __restrict__ s, const bf16_t* __restrict__ w3,
    const float* __restrict__ b3, float* __restrict__ out)
{
  __shared__ __align__(16) bf16_t ldsH[HROWS * HCOLS * HSTR];  // [ext-pixel][c]
  const int tid = threadIdx.x;
  const int lane = tid & 31, wave = tid >> 5;
  const int lrow = lane & 15;
  const int hi8 = (lane & 16) ? 8 : 0;
  const int mtile = wave & 3;
  const int ngrp = wave >> 2;                 // 0/1 -> N-tiles 0..6 / 7..13
  const int n   = blockIdx.x / 14;
  const int h0  = (blockIdx.x % 14) * E3_ROWS;
  const int cb  = blockIdx.y * 64;
  const bf16_t zero_bf = __builtin_bit_cast(bf16_t, (unsigned short)0);

  // ---- halo fill: 22272 elems, 87 iters, incremental index decomposition ----
  {
    int cc = tid % HCOLS;                     // ext col 0..57
    int rest = tid / HCOLS;
    int rr = rest % HROWS;                    // ext row 0..5
    int ch = rest / HROWS;                    // channel 0..63
    const size_t sbase = (size_t)n * C_SQ * HW;
    #pragma unroll 1
    for (int i = 0; i < 87; ++i) {
      int h = h0 - 1 + rr;
      int w = cc - 1;
      bf16_t v = zero_bf;
      if ((unsigned)h < (unsigned)IMG_W && (unsigned)w < (unsigned)IMG_W)
        v = s[sbase + (size_t)ch * HW + h * IMG_W + w];
      ldsH[(rr * HCOLS + cc) * HSTR + ch] = v;
      // advance flat index by 256 = 4*58 + 24
      cc += 24; int dr = 4;
      if (cc >= HCOLS) { cc -= HCOLS; dr = 5; }
      rr += dr;
      if (rr >= HROWS) { rr -= HROWS; ch += 1; }
    }
  }
  __syncthreads();

  // ---- per-wave N-tile geometry (hoisted out of tap loop) ----
  int eb[7];      // ext-pixel base index per N-tile
  #pragma unroll
  for (int j = 0; j < 7; ++j) {
    int pl = (ngrp * 7 + j) * 16 + lrow;      // 0..223
    int hl = pl / IMG_W;                      // 0..3
    eb[j] = pl + 2 * hl + (HCOLS + 1);        // (hl+1)*58 + (pl-hl*56) + 1
  }

  v8f acc[7];
  #pragma unroll
  for (int j = 0; j < 7; ++j)
    #pragma unroll
    for (int e = 0; e < 8; ++e) acc[j][e] = 0.0f;

  for (int t = 0; t < 9; ++t) {
    const int eoff = (t / 3 - 1) * HCOLS + (t % 3 - 1);   // tap shift (uniform)
    #pragma unroll
    for (int kc = 0; kc < 2; ++kc) {
      const int kb = kc * 32;
      v16bf afrag = load_fragA(
          w3 + ((size_t)t * C_EX + cb + mtile * 16 + lrow) * C_SQ + kb, lane);
      // batch B-frag LDS loads (groups of 4 + 3) ahead of WMMA bursts
      v16bf bfr[4];
      #pragma unroll
      for (int j = 0; j < 4; ++j)
        bfr[j] = load_fragB(&ldsH[(eb[j] + eoff) * HSTR + kb], lane);
      #pragma unroll
      for (int j = 0; j < 4; ++j)
        acc[j] = wmma_bf16(afrag, bfr[j], acc[j]);
      #pragma unroll
      for (int j = 0; j < 3; ++j)
        bfr[j] = load_fragB(&ldsH[(eb[4 + j] + eoff) * HSTR + kb], lane);
      #pragma unroll
      for (int j = 0; j < 3; ++j)
        acc[4 + j] = wmma_bf16(afrag, bfr[j], acc[4 + j]);
    }
  }

  // ---- epilogue: pixel offset = h0*56 + pl (no division needed) ----
  #pragma unroll
  for (int j = 0; j < 7; ++j) {
    int pl = (ngrp * 7 + j) * 16 + lrow;
    int pix = h0 * IMG_W + pl;
    #pragma unroll
    for (int r = 0; r < 8; ++r) {
      int cl = cb + mtile * 16 + r + hi8;     // 0..255 within e3
      float v = acc[j][r] + b3[cl];
      v = v > 0.0f ? v : 0.0f;
      out[((size_t)n * 512 + C_EX + cl) * HW + pix] = v;
    }
  }
}

// ---------------------------------------------------------------------------
extern "C" void kernel_launch(void* const* d_in, const int* in_sizes, int n_in,
                              void* d_out, int out_size, void* d_ws, size_t ws_size,
                              hipStream_t stream) {
  const float* x   = (const float*)d_in[0];
  const float* wsq = (const float*)d_in[1];
  const float* bsq = (const float*)d_in[2];
  const float* w1  = (const float*)d_in[3];
  const float* b1  = (const float*)d_in[4];
  const float* w3  = (const float*)d_in[5];
  const float* b3  = (const float*)d_in[6];
  float* out = (float*)d_out;

  char* ws = (char*)d_ws;
  bf16_t* g_wsq = (bf16_t*)(ws);               // 64*512*2   = 65536 B
  bf16_t* g_w1  = (bf16_t*)(ws + 65536);       // 256*64*2   = 32768 B
  bf16_t* g_w3  = (bf16_t*)(ws + 98304);       // 9*256*64*2 = 294912 B
  bf16_t* g_s   = (bf16_t*)(ws + 393216);      // 32*64*3136*2 = 12.8 MB

  const int QTILES = (NIMG * HW) / NT;         // 784

  k_prep<<<dim3(576), dim3(256), 0, stream>>>(wsq, w1, w3, g_wsq, g_w1, g_w3);
  k_squeeze<<<dim3(QTILES), dim3(256), 0, stream>>>(x, g_wsq, bsq, g_s);
  k_expand1<<<dim3(QTILES, 4), dim3(256), 0, stream>>>(g_s, g_w1, b1, out);
  k_expand3<<<dim3(NIMG * 14, 4), dim3(256), 0, stream>>>(g_s, g_w3, b3, out);
}
